// Equilibrium3BLayer_41936060678331
// MI455X (gfx1250) — matrix operand order repaired and checked by reading, not compiled, not run on hardware
//
#include <hip/hip_runtime.h>
#include <hip/hip_bf16.h>
#include <math.h>

typedef __attribute__((ext_vector_type(16))) __bf16 v16bf;
typedef __attribute__((ext_vector_type(8)))  float  v8f;

#define D_MODEL   2560
#define D_INNER   5120
#define N_STATE   128
#define HEADS     80
#define P_HEAD    64
#define CONV_DIM  5376
#define D_IN_PROJ 10576
#define N_EXPERTS 16
#define M_INTER   864
#define MS_INTER  1728
#define SEQ       1024

// ---------------------------------------------------------------------------
// bf16 WMMA GEMM with 4x2 register blocking per wave:
//   C[M,N] = A[M,K](bf16) * B[K,N](fp32, cvt->bf16 in VGPRs)
// Wave tile = 64(M) x 32(N) -> 8 WMMAs per K-step; B frags reused 4x,
// A frags reused 2x (B is the expensive fp32 stream -> reuse it most).
// Block = 256 threads = 8 waves (2M x 4N) -> 128 x 128 block tile.
// Optional: residual add, bf16 store, row-gather on A, scaled atomic row-
// scatter on C, device-side M (dynamic expert token counts).
// No LDS: ragged waves may return early without barrier hazards.
// ---------------------------------------------------------------------------
__global__ __launch_bounds__(256) void wmma_gemm_kernel(
    const __bf16* __restrict__ A, int lda,
    const float*  __restrict__ Bw, int ldb,
    float*        __restrict__ C, int ldc,
    const float*  __restrict__ addC,
    __bf16*       __restrict__ Cbf,
    const int*    __restrict__ gatherA,
    const int*    __restrict__ scatterC,
    const float*  __restrict__ scaleC,
    const int*    __restrict__ Mdev, int Mfixed,
    int N, int K)
{
    int M = Mdev ? *Mdev : Mfixed;
    if ((int)(blockIdx.y * 128) >= M) return;

    int tid  = threadIdx.x;
    int lane = tid & 31;
    int wid  = tid >> 5;
    int m0 = (int)blockIdx.y * 128 + (wid >> 2) * 64;  // 2 waves in M
    int n0 = (int)blockIdx.x * 128 + (wid & 3) * 32;   // 4 waves in N
    if (n0 >= N || m0 >= M) return;                    // wave-uniform exit
    bool nv1 = (n0 + 32 <= N);                         // 2nd N-subtile valid

    // A fragments (16x32 bf16, ISA 7.12.2):
    // lane L: row = L&15 ; K = (L>>4)*8 + {0..7} and +16..+23
    int koff = (lane >> 4) << 3;
    const __bf16* Ap[4];
    #pragma unroll
    for (int mi = 0; mi < 4; ++mi) {
        int r = m0 + mi * 16 + (lane & 15);
        int rc = r < M ? r : (M - 1);
        int ar = gatherA ? gatherA[rc] : rc;
        Ap[mi] = A + (size_t)ar * lda + koff;
    }

    // B fragments (32x16 bf16): lane L: col = L&15 ; K = (L>>4)*8 + {0..7} and +16..+23
    int c0 = n0 + (lane & 15);
    int c1 = c0 + 16;
    if (!nv1) c1 = N - 1;                              // clamp, store masked
    const float* Bp0 = Bw + (size_t)((lane >> 4) << 3) * ldb + c0;
    const float* Bp1 = Bw + (size_t)((lane >> 4) << 3) * ldb + c1;

    v8f acc[4][2] = {};
    for (int kk = 0; kk < K; kk += 32) {
        union UA { v16bf v; uint4 q[2]; } ua[4];
        #pragma unroll
        for (int mi = 0; mi < 4; ++mi) {
            ua[mi].q[0] = *(const uint4*)(Ap[mi] + kk);
            ua[mi].q[1] = *(const uint4*)(Ap[mi] + kk + 16);
        }

        union UB { v16bf v; __bf16 e[16]; } ub0, ub1;
        const float* b0 = Bp0 + (size_t)kk * ldb;
        const float* b1 = Bp1 + (size_t)kk * ldb;
        #pragma unroll
        for (int j = 0; j < 16; ++j) {
            ub0.e[j] = (__bf16)b0[(size_t)(j + (j & 8)) * ldb];
            ub1.e[j] = (__bf16)b1[(size_t)(j + (j & 8)) * ldb];
        }

        #pragma unroll
        for (int mi = 0; mi < 4; ++mi) {
            acc[mi][0] = __builtin_amdgcn_wmma_f32_16x16x32_bf16(
                false, ua[mi].v, false, ub0.v, (short)0, acc[mi][0], false, false);
            acc[mi][1] = __builtin_amdgcn_wmma_f32_16x16x32_bf16(
                false, ua[mi].v, false, ub1.v, (short)0, acc[mi][1], false, false);
        }
    }

    // Epilogue. C layout: VGPR r -> row r (lanes 0-15) / row 8+r (lanes 16-31)
    int nlo = n0 + (lane & 15);
    int rofs = (lane >> 4) << 3;
    #pragma unroll
    for (int mi = 0; mi < 4; ++mi) {
        #pragma unroll
        for (int ni = 0; ni < 2; ++ni) {
            if (ni == 1 && !nv1) continue;
            int n = nlo + ni * 16;
            #pragma unroll
            for (int r = 0; r < 8; ++r) {
                int m = m0 + mi * 16 + r + rofs;
                if (m < M) {
                    float v = acc[mi][ni][r];
                    if (scatterC) {
                        atomicAdd(&C[(size_t)scatterC[m] * ldc + n], scaleC[m] * v);
                    } else if (Cbf) {
                        Cbf[(size_t)m * ldc + n] = (__bf16)v;
                    } else {
                        if (addC) v += addC[(size_t)m * ldc + n];
                        C[(size_t)m * ldc + n] = v;
                    }
                }
            }
        }
    }
}

// ---------------------------------------------------------------------------
// RMSNorm (fp32 in) -> bf16 out, one block per row
// ---------------------------------------------------------------------------
__global__ __launch_bounds__(256) void rmsnorm_bf16_kernel(
    const float* __restrict__ x, const float* __restrict__ w,
    __bf16* __restrict__ out, int d)
{
    int row = blockIdx.x;
    const float* xr = x + (size_t)row * d;
    float ss = 0.f;
    for (int i = threadIdx.x; i < d; i += 256) { float v = xr[i]; ss += v * v; }
    __shared__ float red[256];
    red[threadIdx.x] = ss; __syncthreads();
    for (int s = 128; s > 0; s >>= 1) {
        if ((int)threadIdx.x < s) red[threadIdx.x] += red[threadIdx.x + s];
        __syncthreads();
    }
    float inv = rsqrtf(red[0] / (float)d + 1e-6f);
    for (int i = threadIdx.x; i < d; i += 256)
        out[(size_t)row * d + i] = (__bf16)(w[i] * (xr[i] * inv));
}

// Gated RMSNorm: u = y * silu(z); out = bf16(w * rms(u))
__global__ __launch_bounds__(256) void gated_rmsnorm_kernel(
    const float* __restrict__ y, const float* __restrict__ zf, int ldz,
    const float* __restrict__ w, __bf16* __restrict__ out, int d)
{
    int row = blockIdx.x;
    const float* yr = y + (size_t)row * d;
    const float* zr = zf + (size_t)row * ldz;
    float ss = 0.f;
    for (int i = threadIdx.x; i < d; i += 256) {
        float z = zr[i];
        float u = yr[i] * (z / (1.f + expf(-z)));
        ss += u * u;
    }
    __shared__ float red[256];
    red[threadIdx.x] = ss; __syncthreads();
    for (int s = 128; s > 0; s >>= 1) {
        if ((int)threadIdx.x < s) red[threadIdx.x] += red[threadIdx.x + s];
        __syncthreads();
    }
    float inv = rsqrtf(red[0] / (float)d + 1e-6f);
    for (int i = threadIdx.x; i < d; i += 256) {
        float z = zr[i];
        float u = yr[i] * (z / (1.f + expf(-z)));
        out[(size_t)row * d + i] = (__bf16)(w[i] * (u * inv));
    }
}

// ---------------------------------------------------------------------------
// Depthwise causal conv(4) over xBC channels + bias + SiLU
// ---------------------------------------------------------------------------
__global__ __launch_bounds__(256) void conv_silu_kernel(
    const float* __restrict__ zx, const float* __restrict__ cw,
    const float* __restrict__ cb, float* __restrict__ outp)
{
    int idx = (int)blockIdx.x * 256 + threadIdx.x;
    if (idx >= SEQ * CONV_DIM) return;
    int l = idx / CONV_DIM, c = idx % CONV_DIM;
    float acc = cb[c];
    #pragma unroll
    for (int k = 0; k < 4; ++k) {
        int lt = l - 3 + k;
        if (lt >= 0) acc += zx[(size_t)lt * D_IN_PROJ + D_INNER + c] * cw[c * 4 + k];
    }
    outp[idx] = acc / (1.f + expf(-acc));
}

__global__ __launch_bounds__(256) void dt_softplus_kernel(
    const float* __restrict__ zx, const float* __restrict__ dtb,
    float* __restrict__ outp)
{
    int idx = (int)blockIdx.x * 256 + threadIdx.x;
    if (idx >= SEQ * HEADS) return;
    int l = idx / HEADS, h = idx % HEADS;
    float v = zx[(size_t)l * D_IN_PROJ + (D_INNER + CONV_DIM) + h] + dtb[h];
    outp[idx] = (v > 20.f) ? v : log1pf(expf(v));
}

// ---------------------------------------------------------------------------
// Selective scan: one block per head; state 64x128 fp32 in registers
// (32 per thread); B/C staged in LDS (256 floats = one load per thread).
// ---------------------------------------------------------------------------
__global__ __launch_bounds__(256) void scan_kernel(
    const float* __restrict__ xbc, const float* __restrict__ dtb,
    const float* __restrict__ A_log, const float* __restrict__ Dp,
    float* __restrict__ y)
{
    int h = blockIdx.x;
    int tid = threadIdx.x;
    int p = tid & 63, nb = tid >> 6;     // nb in 0..3, 32 n-values each
    float Ah = -expf(A_log[h]);
    float Dh = Dp[h];
    float hs[32];
    #pragma unroll
    for (int j = 0; j < 32; ++j) hs[j] = 0.f;

    __shared__ float bc[256];            // B[0..127] | C[0..127]
    __shared__ float part[256];          // partial y: [p][nb]

    for (int l = 0; l < SEQ; ++l) {
        const float* row = xbc + (size_t)l * CONV_DIM;
        bc[tid] = row[D_INNER + tid];
        __syncthreads();
        float dt = dtb[l * HEADS + h];
        float dA = expf(dt * Ah);
        float xp = row[h * P_HEAD + p];
        float dx = dt * xp;
        float acc = 0.f;
        #pragma unroll
        for (int j = 0; j < 32; ++j) {
            int n = nb * 32 + j;
            hs[j] = hs[j] * dA + dx * bc[n];
            acc  += hs[j] * bc[128 + n];
        }
        part[p * 4 + nb] = acc;
        __syncthreads();
        if (nb == 0) {
            float s = part[p * 4] + part[p * 4 + 1] + part[p * 4 + 2] + part[p * 4 + 3];
            y[(size_t)l * D_INNER + h * P_HEAD + p] = s + Dh * xp;
        }
        __syncthreads();                 // protect bc/part for next step
    }
}

// ---------------------------------------------------------------------------
// Router: logits -> top-2 (renormalized). One block per token.
// ---------------------------------------------------------------------------
__global__ __launch_bounds__(256) void router_kernel(
    const __bf16* __restrict__ xm, const float* __restrict__ rw,
    int* __restrict__ topi, float* __restrict__ topw)
{
    int t = blockIdx.x;
    int e = threadIdx.x & 15, chunk = threadIdx.x >> 4;
    const __bf16* xr = xm + (size_t)t * D_MODEL;
    float acc = 0.f;
    for (int i = chunk; i < D_MODEL; i += 16)
        acc += (float)xr[i] * rw[(size_t)i * N_EXPERTS + e];
    __shared__ float red[256];
    red[threadIdx.x] = acc; __syncthreads();
    if (threadIdx.x < 16) {
        float s = 0.f;
        for (int c = 0; c < 16; ++c) s += red[c * 16 + threadIdx.x];
        red[threadIdx.x] = s;
    }
    __syncthreads();
    if (threadIdx.x == 0) {
        float mx = red[0];
        for (int i = 1; i < 16; ++i) mx = fmaxf(mx, red[i]);
        float ex[16];
        for (int i = 0; i < 16; ++i) ex[i] = expf(red[i] - mx);
        int i0 = 0;
        for (int i = 1; i < 16; ++i) if (ex[i] > ex[i0]) i0 = i;
        int i1 = (i0 == 0) ? 1 : 0;
        for (int i = 0; i < 16; ++i) if (i != i0 && ex[i] > ex[i1]) i1 = i;
        float p0 = ex[i0], p1 = ex[i1], sum2 = p0 + p1;
        topi[t * 2] = i0; topi[t * 2 + 1] = i1;
        topw[t * 2] = p0 / sum2; topw[t * 2 + 1] = p1 / sum2;
    }
}

// Deterministic per-expert token lists (serial scan per expert).
__global__ void build_lists_kernel(
    const int* __restrict__ topi, const float* __restrict__ topw,
    int* __restrict__ counts, int* __restrict__ list, float* __restrict__ wlist)
{
    if (threadIdx.x != 0) return;
    int e = blockIdx.x;
    int c = 0;
    for (int t = 0; t < SEQ; ++t)
        for (int k = 0; k < 2; ++k)
            if (topi[t * 2 + k] == e) {
                list[e * SEQ + c] = t;
                wlist[e * SEQ + c] = topw[t * 2 + k];
                ++c;
            }
    counts[e] = c;
}

__global__ __launch_bounds__(256) void swiglu_bf_kernel(
    const __bf16* __restrict__ g, const __bf16* __restrict__ u,
    __bf16* __restrict__ o, size_t n)
{
    size_t i = (size_t)blockIdx.x * 256 + threadIdx.x;
    if (i >= n) return;
    float gv = (float)g[i];
    o[i] = (__bf16)((gv / (1.f + expf(-gv))) * (float)u[i]);
}

__global__ __launch_bounds__(256) void swiglu_f32_kernel(
    const float* __restrict__ g, const float* __restrict__ u,
    __bf16* __restrict__ o, size_t n)
{
    size_t i = (size_t)blockIdx.x * 256 + threadIdx.x;
    if (i >= n) return;
    float gv = g[i];
    o[i] = (__bf16)((gv / (1.f + expf(-gv))) * u[i]);
}

__global__ __launch_bounds__(256) void add_kernel(
    const float* __restrict__ a, const float* __restrict__ b,
    float* __restrict__ o, size_t n)
{
    size_t i = (size_t)blockIdx.x * 256 + threadIdx.x;
    if (i < n) o[i] = a[i] + b[i];
}

// ---------------------------------------------------------------------------
static inline unsigned cdivu(size_t a, size_t b) { return (unsigned)((a + b - 1) / b); }

extern "C" void kernel_launch(void* const* d_in, const int* in_sizes, int n_in,
                              void* d_out, int out_size, void* d_ws, size_t ws_size,
                              hipStream_t stream)
{
    (void)in_sizes; (void)n_in; (void)out_size; (void)ws_size;
    const float* hsin    = (const float*)d_in[0];
    const float* w_in    = (const float*)d_in[1];
    const float* conv_w  = (const float*)d_in[2];
    const float* conv_b  = (const float*)d_in[3];
    const float* dt_bias = (const float*)d_in[4];
    const float* A_log   = (const float*)d_in[5];
    const float* Dp      = (const float*)d_in[6];
    const float* mnw     = (const float*)d_in[7];
    const float* w_out   = (const float*)d_in[8];
    const float* ln1     = (const float*)d_in[9];
    const float* ln2     = (const float*)d_in[10];
    const float* rw      = (const float*)d_in[11];
    const float* weg     = (const float*)d_in[12];
    const float* weu     = (const float*)d_in[13];
    const float* wed     = (const float*)d_in[14];
    const float* wsg     = (const float*)d_in[15];
    const float* wsu     = (const float*)d_in[16];
    const float* wsd     = (const float*)d_in[17];
    float* out = (float*)d_out;

    char* w = (char*)d_ws;
    size_t off = 0;
    auto alloc = [&](size_t b) -> char* {
        char* p = w + off; off += (b + 255) & ~(size_t)255; return p;
    };
    float*  hidden = (float*) alloc((size_t)SEQ * D_MODEL * 4);
    float*  moe    = (float*) alloc((size_t)SEQ * D_MODEL * 4);
    __bf16* xmb    = (__bf16*)alloc((size_t)SEQ * D_MODEL * 2);   // xnorm / xm
    float*  dtb    = (float*) alloc((size_t)SEQ * HEADS * 4);
    int*    topi   = (int*)   alloc((size_t)SEQ * 2 * 4);
    float*  topw   = (float*) alloc((size_t)SEQ * 2 * 4);
    int*    counts = (int*)   alloc(64);
    int*    list   = (int*)   alloc((size_t)N_EXPERTS * SEQ * 4);
    float*  wlist  = (float*) alloc((size_t)N_EXPERTS * SEQ * 4);
    __bf16* y2b    = (__bf16*)alloc((size_t)SEQ * D_INNER * 2);
    // Scratch region S: phase-A (zxbcdt|xBC|y) overlaps phase-B (experts).
    char* S = alloc(103u * 1024u * 1024u);
    float*  zxb  = (float*)(S);                                          // 43.3MB
    float*  xbc  = (float*)(S + (size_t)SEQ * D_IN_PROJ * 4);            // 22.0MB
    float*  ybuf = (float*)(S + (size_t)SEQ * (D_IN_PROJ + CONV_DIM) * 4);// 21.0MB
    size_t EXB = (size_t)N_EXPERTS * SEQ * M_INTER;                      // 14.2M elems
    __bf16* hexg = (__bf16*)(S);
    __bf16* hexu = (__bf16*)(S + EXB * 2);
    __bf16* hex  = (__bf16*)(S + EXB * 4);
    float*  sg   = (float*) (S + EXB * 6);
    float*  su   = (float*) (S + EXB * 6 + (size_t)SEQ * MS_INTER * 4);
    __bf16* hsb  = (__bf16*)(S + EXB * 6 + (size_t)SEQ * MS_INTER * 8);

    auto gemm = [&](const __bf16* A, int lda, const float* B, int ldb,
                    float* C, int ldc, const float* addC, __bf16* Cbf,
                    const int* gA, const int* sC, const float* scC,
                    const int* Mdev, int Mfixed, int N, int K) {
        dim3 grid(cdivu((size_t)N, 128), cdivu((size_t)Mfixed, 128));
        wmma_gemm_kernel<<<grid, 256, 0, stream>>>(A, lda, B, ldb, C, ldc, addC,
                                                   Cbf, gA, sC, scC, Mdev, Mfixed, N, K);
    };

    // ---- Mamba2 sub-block ----
    rmsnorm_bf16_kernel<<<SEQ, 256, 0, stream>>>(hsin, ln1, xmb, D_MODEL);
    gemm(xmb, D_MODEL, w_in, D_IN_PROJ, zxb, D_IN_PROJ,
         nullptr, nullptr, nullptr, nullptr, nullptr, nullptr, SEQ, D_IN_PROJ, D_MODEL);
    conv_silu_kernel<<<cdivu((size_t)SEQ * CONV_DIM, 256), 256, 0, stream>>>(zxb, conv_w, conv_b, xbc);
    dt_softplus_kernel<<<cdivu((size_t)SEQ * HEADS, 256), 256, 0, stream>>>(zxb, dt_bias, dtb);
    scan_kernel<<<HEADS, 256, 0, stream>>>(xbc, dtb, A_log, Dp, ybuf);
    gated_rmsnorm_kernel<<<SEQ, 256, 0, stream>>>(ybuf, zxb, D_IN_PROJ, mnw, y2b, D_INNER);
    gemm(y2b, D_INNER, w_out, D_MODEL, hidden, D_MODEL,
         hsin, nullptr, nullptr, nullptr, nullptr, nullptr, SEQ, D_MODEL, D_INNER);

    // ---- MoE sub-block ----
    rmsnorm_bf16_kernel<<<SEQ, 256, 0, stream>>>(hidden, ln2, xmb, D_MODEL);
    router_kernel<<<SEQ, 256, 0, stream>>>(xmb, rw, topi, topw);
    build_lists_kernel<<<N_EXPERTS, 32, 0, stream>>>(topi, topw, counts, list, wlist);

    // Routed experts: gate/up with A-row gather, dynamic M per expert.
    for (int e = 0; e < N_EXPERTS; ++e) {
        gemm(xmb, D_MODEL, weg + (size_t)e * D_MODEL * M_INTER, M_INTER,
             nullptr, M_INTER, nullptr, hexg + (size_t)e * SEQ * M_INTER,
             list + e * SEQ, nullptr, nullptr, counts + e, SEQ, M_INTER, D_MODEL);
        gemm(xmb, D_MODEL, weu + (size_t)e * D_MODEL * M_INTER, M_INTER,
             nullptr, M_INTER, nullptr, hexu + (size_t)e * SEQ * M_INTER,
             list + e * SEQ, nullptr, nullptr, counts + e, SEQ, M_INTER, D_MODEL);
    }
    swiglu_bf_kernel<<<cdivu(EXB, 256), 256, 0, stream>>>(hexg, hexu, hex, EXB);

    // Shared experts first: plain store initializes moe accumulator.
    gemm(xmb, D_MODEL, wsg, MS_INTER, sg, MS_INTER,
         nullptr, nullptr, nullptr, nullptr, nullptr, nullptr, SEQ, MS_INTER, D_MODEL);
    gemm(xmb, D_MODEL, wsu, MS_INTER, su, MS_INTER,
         nullptr, nullptr, nullptr, nullptr, nullptr, nullptr, SEQ, MS_INTER, D_MODEL);
    swiglu_f32_kernel<<<cdivu((size_t)SEQ * MS_INTER, 256), 256, 0, stream>>>(sg, su, hsb, (size_t)SEQ * MS_INTER);
    gemm(hsb, MS_INTER, wsd, D_MODEL, moe, D_MODEL,
         nullptr, nullptr, nullptr, nullptr, nullptr, nullptr, SEQ, D_MODEL, MS_INTER);

    // Routed down-proj: scaled atomic scatter onto moe accumulator.
    for (int e = 0; e < N_EXPERTS; ++e) {
        gemm(hex + (size_t)e * SEQ * M_INTER, M_INTER,
             wed + (size_t)e * M_INTER * D_MODEL, D_MODEL,
             moe, D_MODEL, nullptr, nullptr, nullptr,
             list + e * SEQ, wlist + e * SEQ, counts + e, SEQ, D_MODEL, M_INTER);
    }

    add_kernel<<<cdivu((size_t)SEQ * D_MODEL, 256), 256, 0, stream>>>(hidden, moe, out, (size_t)SEQ * D_MODEL);
}
